// RNN_Encoder_25323127177658
// MI455X (gfx1250) — matrix operand order repaired and checked by reading, not compiled, not run on hardware
//
#include <hip/hip_runtime.h>
#include <stdint.h>

typedef __attribute__((ext_vector_type(16))) __bf16 v16bf;
typedef __attribute__((ext_vector_type(8)))  float  v8f;

#define T_SEQ 200
#define BATCH 128

// gfx1250 async global->LDS copy path (GLOBAL_LOAD_ASYNC_TO_LDS_B128 + ASYNCcnt).
#define USE_ASYNC_LDS 1

#define KC 64           // K-chunk staged in LDS (2 WMMA k-steps)
#define LDSB_STRIDE 72  // padded bf16 row stride (36 dwords) to spread LDS banks

// ---------------------------------------------------------------------------
// Elementwise helpers
// ---------------------------------------------------------------------------
__global__ void zero_bf16_kernel(unsigned short* p, long n) {
  long i = (long)blockIdx.x * blockDim.x + threadIdx.x;
  if (i < n) p[i] = 0;
}

__global__ void zero_f32_kernel(float* p, long n) {
  long i = (long)blockIdx.x * blockDim.x + threadIdx.x;
  if (i < n) p[i] = 0.f;
}

// Gather embedding rows, convert to bf16, zero-pad K 400 -> 448.
__global__ void embed_cast_kernel(const int* __restrict__ tokens,
                                  const float* __restrict__ emb,
                                  unsigned short* __restrict__ xbf,
                                  int K, int Kpad, long rows) {
  long i = (long)blockIdx.x * blockDim.x + threadIdx.x;
  long n = rows * (long)Kpad;
  if (i >= n) return;
  long row = i / Kpad;
  int  k   = (int)(i - row * Kpad);
  float v = 0.f;
  if (k < K) v = emb[(long)tokens[row] * K + k];
  ((__bf16*)xbf)[i] = (__bf16)v;
}

// f32 [N,K] row-major weight -> bf16 [Npad,Kpad] zero-padded.
__global__ void cast_pad_weight_kernel(const float* __restrict__ W,
                                       unsigned short* __restrict__ Wp,
                                       int N, int K, int Npad, int Kpad) {
  long i = (long)blockIdx.x * blockDim.x + threadIdx.x;
  long n = (long)Npad * Kpad;
  if (i >= n) return;
  int nn = (int)(i / Kpad);
  int kk = (int)(i - (long)nn * Kpad);
  float v = (nn < N && kk < K) ? W[(long)nn * K + kk] : 0.f;
  ((__bf16*)Wp)[i] = (__bf16)v;
}

__global__ void pad_bias_kernel(const float* __restrict__ b,
                                float* __restrict__ bp, int N, int Npad) {
  int i = blockIdx.x * blockDim.x + threadIdx.x;
  if (i < Npad) bp[i] = (i < N) ? b[i] : 0.f;
}

// ---------------------------------------------------------------------------
// bf16 WMMA GEMM:  D[M,Npad] = A[M,Kpad] * B[Npad,Kpad]^T + Cinit
//   bias_mode==1 : Cinit is a row vector [Npad] broadcast over rows
//   bias_mode==0 : Cinit is a full matrix [M,ldci]
// grid: (Npad/64, M/128), block 256 = 8 waves; wave w owns m-tile w; each wave
// computes a 16x64 strip (4 accumulators).
// The 64(N) x 64(K) B chunk is shared by all 8 waves and staged in LDS
// (double-buffered, async-to-LDS fill overlapping the WMMAs).  The A fragments
// are software-pipelined one chunk ahead so their global-load latency is
// covered by the previous chunk's WMMA block.
// ---------------------------------------------------------------------------
__global__ __launch_bounds__(256)
void gemm_bf16_wmma_kernel(const unsigned short* __restrict__ A_, int lda,
                           const unsigned short* __restrict__ B_, int ldb,
                           const float* __restrict__ Cinit, int ldci, int bias_mode,
                           float* __restrict__ D, int ldd, int kChunks) {
  __shared__ alignas(16) __bf16 lbs[2][64 * LDSB_STRIDE];

  const __bf16* A  = (const __bf16*)A_;
  const __bf16* Bm = (const __bf16*)B_;

  const int lane = threadIdx.x & 31;
  const int wave = threadIdx.x >> 5;
  const int half = lane >> 4;   // which 16-lane half of the wave
  const int ln   = lane & 15;

  const int row0 = blockIdx.y * 128 + wave * 16;  // M tile base
  const int n0   = blockIdx.x * 64;               // N strip base

  // ---- accumulator init (bias row or full matrix) ----
  v8f acc[4];
  if (bias_mode) {
#pragma unroll
    for (int nt = 0; nt < 4; ++nt) {
      float bv = Cinit[n0 + nt * 16 + ln];
#pragma unroll
      for (int r = 0; r < 8; ++r) acc[nt][r] = bv;
    }
  } else {
#pragma unroll
    for (int nt = 0; nt < 4; ++nt) {
      const float* cp = Cinit + (long)(row0 + half * 8) * ldci + n0 + nt * 16 + ln;
#pragma unroll
      for (int r = 0; r < 8; ++r) acc[nt][r] = cp[(long)r * ldci];
    }
  }

  // ---- cooperative B-chunk staging: 64 rows x 64 k (8 KB) per chunk ----
  auto issue_chunk = [&](int buf, int kc) {
    int t = threadIdx.x;
#pragma unroll
    for (int r = 0; r < 2; ++r) {
      int id   = t + r * 256;   // 0..511
      int row  = id >> 3;       // 0..63
      int part = id & 7;        // 8 bf16 (16B) each
      const __bf16* g = Bm + (long)(n0 + row) * ldb + kc + part * 8;
      __bf16* l = &lbs[buf][row * LDSB_STRIDE + part * 8];
#if USE_ASYNC_LDS
      unsigned int laddr = (unsigned int)(uintptr_t)l;
      asm volatile("global_load_async_to_lds_b128 %0, %1, off"
                   :: "v"(laddr), "v"((unsigned long long)(uintptr_t)g)
                   : "memory");
#else
      *(uint4*)l = *(const uint4*)g;
#endif
    }
  };
  auto wait_chunk = [&]() {
#if USE_ASYNC_LDS
    asm volatile("s_wait_asynccnt 0x0" ::: "memory");
#endif
    __syncthreads();
  };

  // A fragment: lane ln holds row (row0+ln); half 0 -> K {0..7,16..23},
  // half 1 -> K {8..15,24..31} relative to k-step base.
  const __bf16* arow = A + (long)(row0 + ln) * lda + half * 8;
  auto load_a = [&](int kbase) -> v16bf {
    v16bf a;
    const __bf16* ap = arow + kbase;
#pragma unroll
    for (int i = 0; i < 8; ++i) { a[i] = ap[i]; a[8 + i] = ap[16 + i]; }
    return a;
  };

  issue_chunk(0, 0);
  v16bf a0 = load_a(0);       // pipelined A fragments (2 k-steps / chunk)
  v16bf a1 = load_a(32);
  wait_chunk();

  for (int c = 0; c < kChunks; ++c) {
    const int cur = c & 1;
    const bool more = (c + 1 < kChunks);
    if (more) issue_chunk(cur ^ 1, (c + 1) * KC);

    v16bf ac0 = a0, ac1 = a1;
    if (more) {                 // prefetch next chunk's A while computing
      a0 = load_a((c + 1) * KC);
      a1 = load_a((c + 1) * KC + 32);
    }

    const __bf16* lb = lbs[cur];
#pragma unroll
    for (int ks = 0; ks < 2; ++ks) {
      // load all 4 B fragments first so the DS waits batch, then 4 WMMAs
      v16bf bfr[4];
#pragma unroll
      for (int nt = 0; nt < 4; ++nt) {
        const __bf16* bp = lb + (nt * 16 + ln) * LDSB_STRIDE + ks * 32 + half * 8;
#pragma unroll
        for (int i = 0; i < 8; ++i) { bfr[nt][i] = bp[i]; bfr[nt][8 + i] = bp[16 + i]; }
      }
      v16bf av = ks ? ac1 : ac0;
#pragma unroll
      for (int nt = 0; nt < 4; ++nt) {
        acc[nt] = __builtin_amdgcn_wmma_f32_16x16x32_bf16(
            false, av, false, bfr[nt], (short)0, acc[nt], false, false);
      }
    }
    if (more) wait_chunk();
  }

  // ---- store D: VGPR r -> row (r + half*8), column n0+nt*16+ln ----
#pragma unroll
  for (int nt = 0; nt < 4; ++nt) {
    float* dp = D + (long)(row0 + half * 8) * ldd + n0 + nt * 16 + ln;
#pragma unroll
    for (int r = 0; r < 8; ++r) dp[(long)r * ldd] = acc[nt][r];
  }
}

// ---------------------------------------------------------------------------
// LSTM gates (PyTorch order i,f,g,o): consumes g[BATCH, ldg] (= xg_t + h W_hh^T),
// updates c, writes h to d_out (f32), the bf16 step buffer, and (optionally)
// the next layer's bf16 input matrix.
// ---------------------------------------------------------------------------
__global__ void lstm_gate_kernel(const float* __restrict__ g, int ldg, int H, int Hpad,
                                 float* __restrict__ c,
                                 unsigned short* __restrict__ hbf,
                                 unsigned short* __restrict__ xnext, int ldxn,
                                 float* __restrict__ hout, long n) {
  long i = (long)blockIdx.x * blockDim.x + threadIdx.x;
  if (i >= n) return;
  int b = (int)(i / H);
  int j = (int)(i - (long)b * H);
  const float* gr = g + (long)b * ldg;
  float gi = gr[j];
  float gf = gr[H + j];
  float gg = gr[2 * H + j];
  float go = gr[3 * H + j];
  float si = 1.f / (1.f + __expf(-gi));
  float sf = 1.f / (1.f + __expf(-gf));
  float so = 1.f / (1.f + __expf(-go));
  float cn = sf * c[(long)b * Hpad + j] + si * tanhf(gg);
  c[(long)b * Hpad + j] = cn;
  float h = so * tanhf(cn);
  hout[i] = h;
  __bf16 hb = (__bf16)h;
  ((__bf16*)hbf)[(long)b * Hpad + j] = hb;
  if (xnext) ((__bf16*)xnext)[(long)b * ldxn + j] = hb;
}

// ---------------------------------------------------------------------------
// Host orchestration
// ---------------------------------------------------------------------------
static inline long cdiv(long a, long b) { return (a + b - 1) / b; }

extern "C" void kernel_launch(void* const* d_in, const int* in_sizes, int n_in,
                              void* d_out, int out_size, void* d_ws, size_t ws_size,
                              hipStream_t stream) {
  (void)in_sizes; (void)n_in; (void)out_size; (void)ws_size;

  const int*   tokens = (const int*)d_in[0];
  const float* emb    = (const float*)d_in[1];
  const float* Wih[3] = {(const float*)d_in[2], (const float*)d_in[5], (const float*)d_in[8]};
  const float* Whh[3] = {(const float*)d_in[3], (const float*)d_in[6], (const float*)d_in[9]};
  const float* bias[3]= {(const float*)d_in[4], (const float*)d_in[7], (const float*)d_in[10]};

  // layer dims (padded: K to mult of 64 for LDS chunking, N to mult of 64)
  const int Kin[3]    = {400, 1150, 1150};
  const int KinPad[3] = {448, 1152, 1152};
  const int H[3]      = {1150, 1150, 400};
  const int Hpad[3]   = {1152, 1152, 448};
  const int N4[3]     = {4600, 4600, 1600};
  const int Npad[3]   = {4608, 4608, 1600};
  const long ROWS = (long)T_SEQ * BATCH;  // 25600

  // workspace carve-out
  char* ws = (char*)d_ws;
  size_t off = 0;
  auto carve = [&](size_t bytes) -> char* {
    char* p = ws + off;
    off += (bytes + 255) & ~(size_t)255;
    return p;
  };
  float*          ws_xg  = (float*)         carve((size_t)ROWS * 4608 * 4);  // xg, f32
  float*          ws_g   = (float*)         carve((size_t)BATCH * 4608 * 4); // per-step gates
  unsigned short* ws_xA  = (unsigned short*)carve((size_t)ROWS * 1152 * 2);  // layer input (bf16)
  unsigned short* ws_xB  = (unsigned short*)carve((size_t)ROWS * 1152 * 2);  // next layer input
  unsigned short* ws_h   = (unsigned short*)carve((size_t)BATCH * 1152 * 2); // h_t (bf16)
  float*          ws_c   = (float*)         carve((size_t)BATCH * 1152 * 4); // c_t (f32)
  unsigned short* ws_Wih = (unsigned short*)carve((size_t)4608 * 1152 * 2);
  unsigned short* ws_Whh = (unsigned short*)carve((size_t)4608 * 1152 * 2);
  float*          ws_b   = (float*)         carve((size_t)4608 * 4);

  // embedding gather -> bf16 padded [25600, 448]
  {
    long n = ROWS * 448;
    embed_cast_kernel<<<cdiv(n, 256), 256, 0, stream>>>(tokens, emb, ws_xA, 400, 448, ROWS);
  }

  float* out = (float*)d_out;
  size_t outOff = 0;
  unsigned short* xcur = ws_xA;
  unsigned short* xnext = ws_xB;

  for (int l = 0; l < 3; ++l) {
    // cast + pad weights / bias for this layer
    {
      long n = (long)Npad[l] * KinPad[l];
      cast_pad_weight_kernel<<<cdiv(n, 256), 256, 0, stream>>>(
          Wih[l], ws_Wih, N4[l], Kin[l], Npad[l], KinPad[l]);
    }
    {
      long n = (long)Npad[l] * Hpad[l];
      cast_pad_weight_kernel<<<cdiv(n, 256), 256, 0, stream>>>(
          Whh[l], ws_Whh, N4[l], H[l], Npad[l], Hpad[l]);
    }
    pad_bias_kernel<<<cdiv(4608, 256), 256, 0, stream>>>(bias[l], ws_b, N4[l], Npad[l]);

    // xg = x * W_ih^T + b  for all T at once (M=25600)
    {
      dim3 grid(Npad[l] / 64, (unsigned)(ROWS / 128));
      gemm_bf16_wmma_kernel<<<grid, 256, 0, stream>>>(
          xcur, KinPad[l], ws_Wih, KinPad[l],
          ws_b, 0, /*bias_mode=*/1, ws_xg, Npad[l], KinPad[l] / KC);
    }

    // zero recurrent state (h, c) and the next layer's padded input
    zero_bf16_kernel<<<cdiv((long)BATCH * 1152, 256), 256, 0, stream>>>(ws_h, (long)BATCH * 1152);
    zero_f32_kernel<<<cdiv((long)BATCH * 1152, 256), 256, 0, stream>>>(ws_c, (long)BATCH * 1152);
    if (l < 2) {
      long n = ROWS * 1152;
      zero_bf16_kernel<<<cdiv(n, 256), 256, 0, stream>>>(xnext, n);
    }

    // sequential scan: g_t = xg_t + h_{t-1} W_hh^T ; gates ; h_t
    for (int t = 0; t < T_SEQ; ++t) {
      dim3 grid(Npad[l] / 64, 1);
      gemm_bf16_wmma_kernel<<<grid, 256, 0, stream>>>(
          ws_h, Hpad[l], ws_Whh, Hpad[l],
          ws_xg + (size_t)t * BATCH * Npad[l], Npad[l], /*bias_mode=*/0,
          ws_g, Npad[l], Hpad[l] / KC);

      long n = (long)BATCH * H[l];
      lstm_gate_kernel<<<cdiv(n, 256), 256, 0, stream>>>(
          ws_g, Npad[l], H[l], Hpad[l], ws_c, ws_h,
          (l < 2) ? (xnext + (size_t)t * BATCH * 1152) : (unsigned short*)nullptr, 1152,
          out + outOff + (size_t)t * BATCH * H[l], n);
    }

    outOff += (size_t)T_SEQ * BATCH * H[l];
    unsigned short* tmp = xcur; xcur = xnext; xnext = tmp;
  }
}
